// Transformer_33895881900368
// MI455X (gfx1250) — matrix operand order repaired and checked by reading, not compiled, not run on hardware
//
#include <hip/hip_runtime.h>
#include <hip/hip_bf16.h>

// ---------------------------------------------------------------------------
// Transformer block forward for MI455X (gfx1250), bf16 WMMA everywhere.
// B=4, S=2048, F=256, H=8, D=32.  v = q (reference bug kept).
// GEMM/projection B-tiles staged in LDS with gfx1250 async global->LDS copies
// (ASYNCcnt / s_wait_asynccnt), double-buffered, shared by all 4 waves.
// ---------------------------------------------------------------------------

typedef __bf16 bf16;
typedef bf16  v16bf __attribute__((ext_vector_type(16)));
typedef bf16  v8bf  __attribute__((ext_vector_type(8)));
typedef float v8f   __attribute__((ext_vector_type(8)));

#define TB 4
#define TS 2048
#define TF 256
#define TH 8
#define TD 32
#define THD 256      // H*D
#define TF4 1024     // 4*F
#define TBS 8192     // B*S
#define TSF 524288   // S*F
#define NCHUNK 256

#define WMMA_BF16(a, b, c) \
  __builtin_amdgcn_wmma_f32_16x16x32_bf16(false, (a), false, (b), (short)0, (c), false, false)

// ---- gfx1250 async global->LDS copy (16 bytes per lane) -------------------
__device__ __forceinline__ void async_copy16(const bf16* gsrc, bf16* ldst) {
  unsigned lds_addr = (unsigned)(unsigned long long)ldst;  // LDS aperture: low 32 bits
  asm volatile("global_load_async_to_lds_b128 %0, %1, off"
               :: "v"(lds_addr), "v"(gsrc)
               : "memory");
}
__device__ __forceinline__ void wait_async0() {
  asm volatile("s_wait_asynccnt 0x0" ::: "memory");
}

// ---- fragment loaders -----------------------------------------------------
__device__ __forceinline__ v16bf frag_join(const bf16* p0, const bf16* p1) {
  v8bf lo = *(const v8bf*)p0;
  v8bf hi = *(const v8bf*)p1;
  return __builtin_shufflevector(lo, hi, 0,1,2,3,4,5,6,7,8,9,10,11,12,13,14,15);
}

// A fragment: 16x32 row-major tile at (row0, k0), leading dim lda.
// lane<16: row=lane, K chunks {k0+0..7, k0+16..23}; lane>=16: row=lane-16, {k0+8..15, k0+24..31}
__device__ __forceinline__ v16bf load_a_frag(const bf16* A, int lda, int row0, int k0) {
  const int lane = threadIdx.x & 31;
  const bf16* p = A + (size_t)(row0 + (lane & 15)) * lda + k0 + ((lane >> 4) << 3);
  return frag_join(p, p + 16);
}

// B fragment from BT (N x K row-major): lane holds column n0+(lane&15),
// lane<16: K chunks {k0+0..7, k0+16..23}; lane>=16: {k0+8..15, k0+24..31}.
__device__ __forceinline__ v16bf load_b_frag(const bf16* BT, int ldb, int n0, int k0) {
  const int lane = threadIdx.x & 31;
  const bf16* p = BT + (size_t)(n0 + (lane & 15)) * ldb + k0 + ((lane >> 4) << 3);
  return frag_join(p, p + 16);
}

// reductions across a 16-lane half (masks < 16 keep bit4 fixed)
__device__ __forceinline__ float hmax16(float x) {
  x = fmaxf(x, __shfl_xor(x, 1, 32));
  x = fmaxf(x, __shfl_xor(x, 2, 32));
  x = fmaxf(x, __shfl_xor(x, 4, 32));
  x = fmaxf(x, __shfl_xor(x, 8, 32));
  return x;
}
__device__ __forceinline__ float hsum16(float x) {
  x += __shfl_xor(x, 1, 32);
  x += __shfl_xor(x, 2, 32);
  x += __shfl_xor(x, 4, 32);
  x += __shfl_xor(x, 8, 32);
  return x;
}

// ---- elementwise conversion / weight transpose ----------------------------
__global__ void cvt_f32_bf16_kernel(const float* __restrict__ in, bf16* __restrict__ out, size_t n) {
  size_t i = (size_t)blockIdx.x * blockDim.x + threadIdx.x;
  if (i < n) out[i] = (bf16)in[i];
}

// in: (batch, K, N) fp32 -> out: (batch, N, K) bf16
__global__ void wtrans_kernel(const float* __restrict__ in, bf16* __restrict__ out,
                              int batch, int K, int N) {
  size_t i = (size_t)blockIdx.x * blockDim.x + threadIdx.x;
  size_t total = (size_t)batch * K * N;
  if (i >= total) return;
  int b = (int)(i / ((size_t)K * N));
  int rem = (int)(i % ((size_t)K * N));
  int k = rem / N, n = rem % N;
  out[((size_t)b * N + n) * K + k] = (bf16)in[i];
}

// ---- Q/K projection: q = x@Wq+bq, k = x@Wk+bk ------------------------------
// 4 waves/block, all on same (b,h): Wq/Wk K-step tiles (32x32 each) staged in
// LDS by async copies, double-buffered, shared across the block.
__global__ void proj_qk_kernel(const bf16* __restrict__ xbf,
                               const bf16* __restrict__ WqT, const bf16* __restrict__ WkT,
                               const float* __restrict__ bq, const float* __restrict__ bk,
                               bf16* __restrict__ q, bf16* __restrict__ qT,
                               bf16* __restrict__ kout) {
  __shared__ __align__(16) bf16 wtile[2][2][32 * 32];
  const int tid = threadIdx.x;
  const int lane = tid & 31;
  const int wid = blockIdx.x * 4 + (tid >> 5);
  const int qt = wid & (TS / 16 - 1);   // 0..127
  const int bh = wid >> 7;              // 0..31 (same for all 4 waves)
  const int b = bh >> 3, h = bh & 7;
  const bf16* xb = xbf + (size_t)b * TS * TF;
  const bf16* wq = WqT + (size_t)h * TD * TF;
  const bf16* wk = WkT + (size_t)h * TD * TF;
  const int m0 = qt * 16;
  const int steps = TF / 32;

  auto stage = [&](int buf, int k0) {
    int f = tid * 8;  // 128 threads x 8 elems = 1024 = 32x32 tile
    async_copy16(wq + (size_t)(f >> 5) * TF + k0 + (f & 31), &wtile[buf][0][f]);
    async_copy16(wk + (size_t)(f >> 5) * TF + k0 + (f & 31), &wtile[buf][1][f]);
  };

  v8f cq0 = {}, cq1 = {}, ck0 = {}, ck1 = {};
  stage(0, 0);
  v16bf a = load_a_frag(xb, TF, m0, 0);
  for (int s = 0; s < steps; ++s) {
    wait_async0();
    __syncthreads();
    if (s + 1 < steps) stage((s + 1) & 1, (s + 1) * 32);
    v16bf an = a;
    if (s + 1 < steps) an = load_a_frag(xb, TF, m0, (s + 1) * 32);
    const bf16* tq = wtile[s & 1][0];
    const bf16* tk = wtile[s & 1][1];
    cq0 = WMMA_BF16(a, load_b_frag(tq, 32, 0, 0), cq0);
    cq1 = WMMA_BF16(a, load_b_frag(tq, 32, 16, 0), cq1);
    ck0 = WMMA_BF16(a, load_b_frag(tk, 32, 0, 0), ck0);
    ck1 = WMMA_BF16(a, load_b_frag(tk, 32, 16, 0), ck1);
    a = an;
  }

  const int half = lane >> 4, col = lane & 15;
  const size_t qbase = (size_t)bh * TS * TD;
  const size_t tbase = (size_t)bh * TD * TS;
  const float bq0 = bq[h * TD + col],      bq1 = bq[h * TD + 16 + col];
  const float bk0 = bk[h * TD + col],      bk1 = bk[h * TD + 16 + col];
#pragma unroll
  for (int v = 0; v < 8; ++v) {
    int r = m0 + v + 8 * half;
    float qv0 = cq0[v] + bq0, qv1 = cq1[v] + bq1;
    float kv0 = ck0[v] + bk0, kv1 = ck1[v] + bk1;
    q[qbase + (size_t)r * TD + col]         = (bf16)qv0;
    q[qbase + (size_t)r * TD + 16 + col]    = (bf16)qv1;
    qT[tbase + (size_t)col * TS + r]        = (bf16)qv0;
    qT[tbase + (size_t)(16 + col) * TS + r] = (bf16)qv1;
    kout[qbase + (size_t)r * TD + col]      = (bf16)kv0;
    kout[qbase + (size_t)r * TD + 16 + col] = (bf16)kv1;
  }
}

// ---- flash attention (v = q); one wave per 16-query tile per (b,h) --------
__global__ void attn_kernel(const bf16* __restrict__ q, const bf16* __restrict__ qT,
                            const bf16* __restrict__ kmat, bf16* __restrict__ attn_out) {
  __shared__ __align__(16) bf16 pbuf[4][16 * 32];
  const int lane = threadIdx.x & 31;
  const int wv = threadIdx.x >> 5;
  const int wid = blockIdx.x * 4 + wv;
  const int qt = wid & 127;
  const int bh = wid >> 7;
  const int b = bh >> 3, h = bh & 7;
  const bf16* qb  = q    + (size_t)bh * TS * TD;
  const bf16* kb  = kmat + (size_t)bh * TS * TD;
  const bf16* vtb = qT   + (size_t)bh * TD * TS;   // V^T (D x S), since v = q
  bf16* plds = pbuf[wv];
  const int half = lane >> 4, col = lane & 15;

  v16bf aq = load_a_frag(qb, TD, qt * 16, 0);
  v8f acc0 = {}, acc1 = {};
  float rmax[8], rsum[8];
#pragma unroll
  for (int v = 0; v < 8; ++v) { rmax[v] = -3.0e38f; rsum[v] = 0.0f; }
  const float sc = 0.17677669529663687f;   // 1/sqrt(32)

  // pipeline: K-tile score fragments loaded one 2-tile step ahead
  v16bf fk0 = load_b_frag(kb, TD, 0, 0);
  v16bf fk1 = load_b_frag(kb, TD, 16, 0);
  for (int kt = 0; kt < TS / 16; kt += 2) {
    v8f s0 = {}, s1 = {};
    s0 = WMMA_BF16(aq, fk0, s0);
    s1 = WMMA_BF16(aq, fk1, s1);
    // issue V fragments + next K fragments now; they fly during softmax VALU
    v16bf fv0 = load_b_frag(vtb, TS, 0,  kt * 16);
    v16bf fv1 = load_b_frag(vtb, TS, 16, kt * 16);
    if (kt + 2 < TS / 16) {
      fk0 = load_b_frag(kb, TD, (kt + 2) * 16, 0);
      fk1 = load_b_frag(kb, TD, (kt + 3) * 16, 0);
      __builtin_prefetch(kb + (size_t)(kt + 4) * 16 * TD, 0, 0);
    }
#pragma unroll
    for (int v = 0; v < 8; ++v) {
      float a0 = s0[v] * sc, a1 = s1[v] * sc;
      float tmax = hmax16(fmaxf(a0, a1));
      float mnew = fmaxf(rmax[v], tmax);
      float alpha = __expf(rmax[v] - mnew);
      float p0 = __expf(a0 - mnew), p1 = __expf(a1 - mnew);
      rsum[v] = rsum[v] * alpha + hsum16(p0 + p1);
      rmax[v] = mnew;
      acc0[v] *= alpha; acc1[v] *= alpha;
      int r = v + 8 * half;
      plds[r * 32 + col]      = (bf16)p0;
      plds[r * 32 + 16 + col] = (bf16)p1;
    }
    v16bf ap = load_a_frag(plds, 32, 0, 0);
    acc0 = WMMA_BF16(ap, fv0, acc0);
    acc1 = WMMA_BF16(ap, fv1, acc1);
  }
  const size_t obase = (size_t)b * TS * THD + (size_t)h * TD;
#pragma unroll
  for (int v = 0; v < 8; ++v) {
    int r = qt * 16 + v + 8 * half;
    float inv = 1.0f / rsum[v];
    attn_out[obase + (size_t)r * THD + col]      = (bf16)(acc0[v] * inv);
    attn_out[obase + (size_t)r * THD + 16 + col] = (bf16)(acc1[v] * inv);
  }
}

// ---- generic bf16 GEMM: C = A(MxK) * BT(NxK)^T + bias [+res][relu] --------
// Block = 4 waves sharing one 64-col B panel; B K-step tile (64x32) staged in
// LDS with async copies, double-buffered.
__global__ void gemm_kernel(const bf16* __restrict__ A, const bf16* __restrict__ BT,
                            const float* __restrict__ bias, const float* __restrict__ residual,
                            float* __restrict__ outf, bf16* __restrict__ outb,
                            int M, int N, int K, int doRelu) {
  __shared__ __align__(16) bf16 btile[2][64 * 32];
  const int tid = threadIdx.x;
  const int lane = tid & 31;
  const int wv = tid >> 5;
  const int m0 = (blockIdx.y * 4 + wv) * 16;
  const int n0 = blockIdx.x * 64;
  const int steps = K / 32;

  auto stage = [&](int buf, int k0) {
    int f0 = tid * 8, f1 = (tid + 128) * 8;  // 2 x 16B per thread = 64x32 tile
    async_copy16(BT + (size_t)(n0 + (f0 >> 5)) * K + k0 + (f0 & 31), &btile[buf][f0]);
    async_copy16(BT + (size_t)(n0 + (f1 >> 5)) * K + k0 + (f1 & 31), &btile[buf][f1]);
  };

  v8f c0 = {}, c1 = {}, c2 = {}, c3 = {};
  stage(0, 0);
  v16bf a = load_a_frag(A, K, m0, 0);
  for (int s = 0; s < steps; ++s) {
    wait_async0();
    __syncthreads();
    if (s + 1 < steps) stage((s + 1) & 1, (s + 1) * 32);
    v16bf an = a;
    if (s + 1 < steps) an = load_a_frag(A, K, m0, (s + 1) * 32);
    const bf16* bt = btile[s & 1];
    c0 = WMMA_BF16(a, load_b_frag(bt, 32, 0,  0), c0);
    c1 = WMMA_BF16(a, load_b_frag(bt, 32, 16, 0), c1);
    c2 = WMMA_BF16(a, load_b_frag(bt, 32, 32, 0), c2);
    c3 = WMMA_BF16(a, load_b_frag(bt, 32, 48, 0), c3);
    a = an;
  }

  const int half = lane >> 4, col0 = lane & 15;
  v8f* cs[4] = { &c0, &c1, &c2, &c3 };
#pragma unroll
  for (int j = 0; j < 4; ++j) {
    int n = n0 + 16 * j + col0;
    float bv = bias ? bias[n] : 0.0f;
#pragma unroll
    for (int v = 0; v < 8; ++v) {
      int r = m0 + v + 8 * half;
      float val = (*cs[j])[v] + bv;
      if (residual) val += residual[(size_t)r * N + n];
      if (doRelu) val = fmaxf(val, 0.0f);
      if (outf) outf[(size_t)r * N + n] = val;
      if (outb) outb[(size_t)r * N + n] = (bf16)val;
    }
  }
}

// ---- LayerNorm over entire (S,F) slab per batch ---------------------------
__global__ void ln_part_kernel(const float* __restrict__ src, float2* __restrict__ part) {
  __shared__ float ssum[256], ssq[256];
  const int b = blockIdx.y, chunk = blockIdx.x, tid = threadIdx.x;
  const int n = TSF / NCHUNK;
  const float* p = src + (size_t)b * TSF + (size_t)chunk * n;
  float s = 0.0f, sq = 0.0f;
  for (int i = tid; i < n; i += 256) { float x = p[i]; s += x; sq += x * x; }
  ssum[tid] = s; ssq[tid] = sq;
  __syncthreads();
  for (int off = 128; off > 0; off >>= 1) {
    if (tid < off) { ssum[tid] += ssum[tid + off]; ssq[tid] += ssq[tid + off]; }
    __syncthreads();
  }
  if (tid == 0) part[b * NCHUNK + chunk] = make_float2(ssum[0], ssq[0]);
}

__global__ void ln_stats_kernel(const float2* __restrict__ part, float2* __restrict__ stats) {
  __shared__ float ssum[256], ssq[256];
  const int b = blockIdx.x, tid = threadIdx.x;
  float2 p = part[b * NCHUNK + tid];
  ssum[tid] = p.x; ssq[tid] = p.y;
  __syncthreads();
  for (int off = 128; off > 0; off >>= 1) {
    if (tid < off) { ssum[tid] += ssum[tid + off]; ssq[tid] += ssq[tid + off]; }
    __syncthreads();
  }
  if (tid == 0) {
    float mean = ssum[0] / (float)TSF;
    float var = ssq[0] / (float)TSF - mean * mean;
    stats[b] = make_float2(mean, rsqrtf(var + 1e-5f));
  }
}

__global__ void ln_apply_kernel(const float* __restrict__ src, const float2* __restrict__ stats,
                                const float* __restrict__ g, const float* __restrict__ bt,
                                float* __restrict__ outf, bf16* __restrict__ outb) {
  size_t i = (size_t)blockIdx.x * blockDim.x + threadIdx.x;
  if (i >= (size_t)TB * TSF) return;
  int b = (int)(i / TSF);
  int sf = (int)(i % TSF);
  float2 st = stats[b];
  float y = (src[i] - st.x) * st.y * g[sf] + bt[sf];
  if (outf) outf[i] = y;
  if (outb) outb[i] = (bf16)y;
}

// ---------------------------------------------------------------------------
extern "C" void kernel_launch(void* const* d_in, const int* in_sizes, int n_in,
                              void* d_out, int out_size, void* d_ws, size_t ws_size,
                              hipStream_t stream) {
  (void)in_sizes; (void)n_in; (void)out_size; (void)ws_size;
  const float* x     = (const float*)d_in[0];
  const float* Wq    = (const float*)d_in[1];
  const float* bq    = (const float*)d_in[2];
  const float* Wk    = (const float*)d_in[3];
  const float* bk    = (const float*)d_in[4];
  // d_in[5] = Wv, d_in[6] = bv : unused (reference bug kept, v = q)
  const float* Wo    = (const float*)d_in[7];
  const float* bo    = (const float*)d_in[8];
  const float* ln1_g = (const float*)d_in[9];
  const float* ln1_b = (const float*)d_in[10];
  const float* ln2_g = (const float*)d_in[11];
  const float* ln2_b = (const float*)d_in[12];
  const float* W1    = (const float*)d_in[13];
  const float* b1    = (const float*)d_in[14];
  const float* W2    = (const float*)d_in[15];
  const float* b2    = (const float*)d_in[16];
  float* out = (float*)d_out;

  // workspace bump allocator
  char* wsp = (char*)d_ws;
  size_t off = 0;
  auto alloc = [&](size_t bytes) -> void* {
    void* p = wsp + off;
    off = (off + bytes + 255) & ~(size_t)255;
    return p;
  };
  const size_t NQK = (size_t)TB * TH * TS * TD;  // 2M elems
  bf16*  xbf  = (bf16*)alloc((size_t)TBS * TF * 2);
  bf16*  WqT  = (bf16*)alloc((size_t)TH * TD * TF * 2);
  bf16*  WkT  = (bf16*)alloc((size_t)TH * TD * TF * 2);
  bf16*  WoT  = (bf16*)alloc((size_t)THD * TF * 2);
  bf16*  W1T  = (bf16*)alloc((size_t)TF * TF4 * 2);
  bf16*  W2T  = (bf16*)alloc((size_t)TF4 * TF * 2);
  bf16*  qbuf = (bf16*)alloc(NQK * 2);
  bf16*  qTb  = (bf16*)alloc(NQK * 2);
  bf16*  kbuf = (bf16*)alloc(NQK * 2);
  bf16*  atto = (bf16*)alloc((size_t)TBS * THD * 2);
  float* h1   = (float*)alloc((size_t)TBS * TF * 4);
  float* zf   = (float*)alloc((size_t)TBS * TF * 4);
  bf16*  zb   = (bf16*)alloc((size_t)TBS * TF * 2);
  bf16*  ffn1 = (bf16*)alloc((size_t)TBS * TF4 * 2);
  float* h2   = (float*)alloc((size_t)TBS * TF * 4);
  float2* part  = (float2*)alloc((size_t)TB * NCHUNK * sizeof(float2));
  float2* stats = (float2*)alloc((size_t)TB * sizeof(float2));

  // 1) conversions / weight transposes
  cvt_f32_bf16_kernel<<<(TBS * TF) / 256, 256, 0, stream>>>(x, xbf, (size_t)TBS * TF);
  wtrans_kernel<<<(TH * TF * TD) / 256, 256, 0, stream>>>(Wq, WqT, TH, TF, TD);
  wtrans_kernel<<<(TH * TF * TD) / 256, 256, 0, stream>>>(Wk, WkT, TH, TF, TD);
  wtrans_kernel<<<(THD * TF) / 256, 256, 0, stream>>>(Wo, WoT, 1, THD, TF);
  wtrans_kernel<<<(TF * TF4) / 256, 256, 0, stream>>>(W1, W1T, 1, TF, TF4);
  wtrans_kernel<<<(TF4 * TF) / 256, 256, 0, stream>>>(W2, W2T, 1, TF4, TF);

  // 2) Q/K projections (v = q)
  proj_qk_kernel<<<(TB * TH * (TS / 16)) / 4, 128, 0, stream>>>(xbf, WqT, WkT, bq, bk,
                                                               qbuf, qTb, kbuf);
  // 3) attention
  attn_kernel<<<(TB * TH * (TS / 16)) / 4, 128, 0, stream>>>(qbuf, qTb, kbuf, atto);

  // 4) output projection + residual x
  gemm_kernel<<<dim3(TF / 64, TBS / 64), 128, 0, stream>>>(atto, WoT, bo, x, h1, nullptr,
                                                           TBS, TF, THD, 0);
  // 5) LayerNorm1 over (S,F) per batch
  ln_part_kernel<<<dim3(NCHUNK, TB), 256, 0, stream>>>(h1, part);
  ln_stats_kernel<<<TB, 256, 0, stream>>>(part, stats);
  ln_apply_kernel<<<(TB * TSF) / 256, 256, 0, stream>>>(h1, stats, ln1_g, ln1_b, zf, zb);

  // 6) FFN
  gemm_kernel<<<dim3(TF4 / 64, TBS / 64), 128, 0, stream>>>(zb, W1T, b1, nullptr, nullptr, ffn1,
                                                            TBS, TF4, TF, 1);
  gemm_kernel<<<dim3(TF / 64, TBS / 64), 128, 0, stream>>>(ffn1, W2T, b2, zf, h2, nullptr,
                                                           TBS, TF, TF4, 0);
  // 7) LayerNorm2 -> d_out
  ln_part_kernel<<<dim3(NCHUNK, TB), 256, 0, stream>>>(h2, part);
  ln_stats_kernel<<<TB, 256, 0, stream>>>(part, stats);
  ln_apply_kernel<<<(TB * TSF) / 256, 256, 0, stream>>>(h2, stats, ln2_g, ln2_b, out, nullptr);
}
